// TBNN_Q_direction_88081189307080
// MI455X (gfx1250) — matrix operand order repaired and checked by reading, not compiled, not run on hardware
//
#include <hip/hip_runtime.h>
#include <math.h>

typedef __attribute__((ext_vector_type(16))) _Float16 v16h;
typedef __attribute__((ext_vector_type(8)))  float    v8f;
typedef __attribute__((ext_vector_type(2)))  _Float16 h2;
typedef __attribute__((ext_vector_type(4)))  unsigned u32x4;
typedef __attribute__((ext_vector_type(8)))  int      i32x8;
typedef __attribute__((ext_vector_type(4)))  int      i32x4;

#if defined(__has_builtin)
#if __has_builtin(__builtin_amdgcn_tensor_load_to_lds)
#define HAVE_TDM 1
#endif
#endif

union uv16 { v16h h; unsigned u[8]; uint4 q[2]; };

// pack two f32 -> two f16 (RTZ) as our h2 type
__device__ __forceinline__ h2 cvt_pk(float a, float b) {
  return __builtin_bit_cast(h2, __builtin_amdgcn_cvt_pkrtz(a, b));
}

// ---------- small 3x3 helpers (row-major, 9 floats) ----------
__device__ __forceinline__ void mm3(const float* a, const float* b, float* c) {
#pragma unroll
  for (int i = 0; i < 3; ++i)
#pragma unroll
    for (int j = 0; j < 3; ++j)
      c[3*i+j] = a[3*i+0]*b[0+j] + a[3*i+1]*b[3+j] + a[3*i+2]*b[6+j];
}
__device__ __forceinline__ float tr3(const float* a) { return a[0] + a[4] + a[8]; }
__device__ __forceinline__ float trprod(const float* a, const float* b) {
  float t = 0.f;
#pragma unroll
  for (int i = 0; i < 3; ++i)
#pragma unroll
    for (int j = 0; j < 3; ++j)
      t += a[3*i+j] * b[3*j+i];
  return t;
}

// ---------- padded layer geometry (bias folded into weight row k == KA) ----------
#define W_HALFS 52224
#define W_BYTES 104448
// LDS layout (bytes)
#define OFF_W  0
#define OFF_X  104448
#define OFF_G  169984
#define SMEM_BYTES 186368
#define XS 128  // activation row stride in halfs

// transposed, padded, f16, bias-folded weight conversion into dst (52224 halfs)
__device__ __forceinline__ void convert_weights(const float* const* WGl, const float* const* BGl,
                                                _Float16* dst, int tid, int stride) {
  const int KP[6]   = {32, 64, 128, 128, 128, 64};
  const int NP[6]   = {64, 128, 128, 128, 64, 16};
  const int KA[6]   = {5, 50, 100, 100, 100, 50};
  const int NA[6]   = {50, 100, 100, 100, 50, 10};
  const int WOFF[6] = {0, 2048, 10240, 26624, 43008, 51200};
#pragma unroll
  for (int l = 0; l < 6; ++l) {
    const int tot = KP[l] * NP[l];
    for (int idx = tid; idx < tot; idx += stride) {
      const int k = idx / NP[l];
      const int n = idx % NP[l];
      float v = 0.f;
      if (n < NA[l]) {
        if (k < KA[l]) v = WGl[l][k * NA[l] + n];
        else if (k == KA[l]) v = BGl[l][n];       // bias row
      }
      dst[WOFF[l] + n * KP[l] + k] = (_Float16)v;  // transposed [N][K]
    }
  }
}

// ---------- one MLP layer on a wave's 32-row tile via WMMA ----------
// A-frag (16-bit A 16x32): lane L: M=L&15, kb=(L>>4)*8; two 16B chunks per k-step
// B-frag (16-bit B 32x16): lane L: N=L&15, kb=(L>>4)*8; two 16B chunks per k-step (same k order as A)
// C/D   : lane L: N=L&15; VGPR r: M = r + (L>>4)*8
template <int KP, int NP, int NA, bool LAST>
__device__ __forceinline__ void mlp_layer(_Float16* __restrict__ x,
                                          const _Float16* __restrict__ Wt,
                                          float* __restrict__ gbuf, int lane) {
  constexpr int KS = KP / 32;
  constexpr int NT = NP / 16;
  const int m  = lane & 15;
  const int hi = lane >> 4;
  const int kb = hi * 8;

  // Preload ALL A fragments (both M-tiles, all K-steps) -> in-place update is safe.
  uv16 A[2][KS];
#pragma unroll
  for (int mt = 0; mt < 2; ++mt) {
    const _Float16* rowp = x + (mt * 16 + m) * XS;
#pragma unroll
    for (int ks = 0; ks < KS; ++ks) {
      A[mt][ks].q[0] = *(const uint4*)(rowp + ks * 32 + kb);
      A[mt][ks].q[1] = *(const uint4*)(rowp + ks * 32 + 16 + kb);
    }
  }

  const h2 leak = {(_Float16)0.1f, (_Float16)0.1f};
#pragma unroll
  for (int nt = 0; nt < NT; ++nt) {
    v8f acc0 = {};      // bias folded in -> inline-0 C operand
    v8f acc1 = {};
    const _Float16* wrow = Wt + (nt * 16 + m) * KP;
#pragma unroll
    for (int ks = 0; ks < KS; ++ks) {
      uv16 Bf;
      Bf.q[0] = *(const uint4*)(wrow + ks * 32 + kb);
      Bf.q[1] = *(const uint4*)(wrow + ks * 32 + 16 + kb);
      acc0 = __builtin_amdgcn_wmma_f32_16x16x32_f16(false, A[0][ks].h, false, Bf.h,
                                                    (short)0, acc0, false, false);
      acc1 = __builtin_amdgcn_wmma_f32_16x16x32_f16(false, A[1][ks].h, false, Bf.h,
                                                    (short)0, acc1, false, false);
    }
#pragma unroll
    for (int r = 0; r < 8; ++r) {
      const int row0 = r + hi * 8;
      if (!LAST) {
        h2 p  = cvt_pk(acc0[r], acc1[r]);               // pack rows r / r+16
        h2 pm = __builtin_elementwise_max(p, p * leak); // packed LeakyReLU(0.1)
        x[(row0)      * XS + nt * 16 + m] = pm.x;
        x[(row0 + 16) * XS + nt * 16 + m] = pm.y;
      } else {
        gbuf[(row0)      * 16 + m] = acc0[r];
        gbuf[(row0 + 16) * 16 + m] = acc1[r];
      }
    }
  }
  if (!LAST) x[lane * XS + NA] = (_Float16)1.0f;  // "1" column feeds next layer's bias row
}

// ---------- prep kernel: fp32 weights -> f16 transposed/padded/bias-folded blob in d_ws ----------
__global__ void prep_weights_kernel(const float* W0, const float* b0, const float* W1, const float* b1,
                                    const float* W2, const float* b2, const float* W3, const float* b3,
                                    const float* W4, const float* b4, const float* W5, const float* b5,
                                    _Float16* wsW) {
  const float* WGl[6] = {W0, W1, W2, W3, W4, W5};
  const float* BGl[6] = {b0, b1, b2, b3, b4, b5};
  convert_weights(WGl, BGl, wsW, blockIdx.x * blockDim.x + threadIdx.x,
                  gridDim.x * blockDim.x);
}

// ---------- main kernel ----------
template <bool FROMWS>
__global__ __launch_bounds__(256, 1)
void tbnn_q_kernel(const float* __restrict__ Sg, const float* __restrict__ Wg,
                   const float* W0, const float* b0, const float* W1, const float* b1,
                   const float* W2, const float* b2, const float* W3, const float* b3,
                   const float* W4, const float* b4, const float* W5, const float* b5,
                   const _Float16* __restrict__ wsW, float* __restrict__ out, int nItems) {
  extern __shared__ char smem[];
  _Float16* Wt = (_Float16*)(smem + OFF_W);
  _Float16* X  = (_Float16*)(smem + OFF_X);
  float*    G  = (float*)   (smem + OFF_G);

  if (FROMWS) {
#ifdef HAVE_TDM
    // Single TDM DMA: whole 104448-B weight blob, 8B elements, 1 x 13056 tile.
    if (threadIdx.x < 32) {  // wave 0 issues (TDM ignores EXEC; gate per-wave)
      const unsigned ldsA = (unsigned)(unsigned long long)(void*)Wt;
      const unsigned long long ga = (unsigned long long)wsW;
      u32x4 g0;
      g0[0] = 1u;                                   // count=1 valid descriptor
      g0[1] = ldsA;                                 // lds_addr
      g0[2] = (unsigned)ga;                         // global_addr[31:0]
      g0[3] = (unsigned)((ga >> 32) & 0x01FFFFFFull) | 0x80000000u;  // addr[56:32] | type=2
      i32x8 g1;
      g1[0] = 3 << 16;                              // data_size = 8B
      g1[1] = (int)((13056u & 0xFFFFu) << 16);      // tensor_dim0 lo16
      g1[2] = (int)(((13056u >> 16) & 0xFFFFu) | (1u << 16)); // dim0 hi16 | tensor_dim1 lo16
      g1[3] = (int)(13056u << 16);                  // tensor_dim1 hi16=0 | tile_dim0
      g1[4] = 1;                                    // tile_dim1 = 1, tile_dim2 = 0
      g1[5] = 13056;                                // tensor_dim0_stride lo32
      g1[6] = 0;                                    // stride hi16 | dim1_stride lo16
      g1[7] = 0;
      i32x4 gz = {0, 0, 0, 0};
#if defined(__clang_major__) && __clang_major__ >= 23
      i32x8 gz8 = {0, 0, 0, 0, 0, 0, 0, 0};
      __builtin_amdgcn_tensor_load_to_lds(g0, g1, gz, gz, gz8, 0);
#else
      __builtin_amdgcn_tensor_load_to_lds(g0, g1, gz, gz, 0);
#endif
      __builtin_amdgcn_s_wait_tensorcnt(0);
    }
#else
    {
      const uint4* src = (const uint4*)wsW;
      uint4*       dst = (uint4*)Wt;
      for (int i = threadIdx.x; i < W_BYTES / 16; i += 256) dst[i] = src[i];
    }
#endif
  } else {
    const float* WGl[6] = {W0, W1, W2, W3, W4, W5};
    const float* BGl[6] = {b0, b1, b2, b3, b4, b5};
    convert_weights(WGl, BGl, Wt, threadIdx.x, 256);
  }
  __syncthreads();

  const int wv   = threadIdx.x >> 5;
  const int lane = threadIdx.x & 31;
  _Float16* xw = X + wv * 32 * XS;
  float*    gw = G + wv * 32 * 16;

  const long item = (long)blockIdx.x * 256 + threadIdx.x;
  const long idx  = item < nItems ? item : 0;

  // ---- per-lane: load s, w; invariants ----
  float s[9], w[9];
#pragma unroll
  for (int i = 0; i < 9; ++i) { s[i] = Sg[idx * 9 + i]; w[i] = Wg[idx * 9 + i]; }

  float s2[9], w2[9], sw[9], ws[9];
  mm3(s, s, s2); mm3(w, w, w2); mm3(s, w, sw); mm3(w, s, ws);
  const float l1 = tr3(s2);
  const float l2 = tr3(w2);
  const float l3 = trprod(s2, s);
  const float l4 = trprod(w2, s);
  const float l5 = trprod(w2, s2);

  // write invariant row: [l1..l5, 1, 0...0] over 32 cols (Kpad0 = 32)
  {
    unsigned* rowu = (unsigned*)(xw + lane * XS);
    rowu[0] = __builtin_bit_cast(unsigned, cvt_pk(l1, l2));
    rowu[1] = __builtin_bit_cast(unsigned, cvt_pk(l3, l4));
    rowu[2] = __builtin_bit_cast(unsigned, cvt_pk(l5, 1.0f));
    rowu[3] = 0u;
    const uint4 z = {0u, 0u, 0u, 0u};
    *(uint4*)(rowu + 4)  = z;
    *(uint4*)(rowu + 8)  = z;
    *(uint4*)(rowu + 12) = z;
  }

  // ---- MLP via WMMA (wave-private tile; DS ops wave-ordered, no barriers needed) ----
  mlp_layer<32,  64,  50, false>(xw, Wt + 0,     gw, lane);
  mlp_layer<64,  128, 100, false>(xw, Wt + 2048,  gw, lane);
  mlp_layer<128, 128, 100, false>(xw, Wt + 10240, gw, lane);
  mlp_layer<128, 128, 100, false>(xw, Wt + 26624, gw, lane);
  mlp_layer<128, 64,  50, false>(xw, Wt + 43008, gw, lane);
  mlp_layer<64,  16,  10, true >(xw, Wt + 51200, gw, lane);

  // ---- gather g, assemble Q = sum g_n T_n ----
  float g[10];
#pragma unroll
  for (int n = 0; n < 10; ++n) g[n] = gw[lane * 16 + n];

  float Q[9];
#pragma unroll
  for (int i = 0; i < 9; ++i)
    Q[i] = g[0] * s[i] + g[1] * (sw[i] - ws[i]) + g[2] * s2[i] + g[3] * w2[i];
  float diag = g[2] * (l1 * (1.f / 3.f)) + g[3] * (l2 * (1.f / 3.f));

  float ws2[9], s2w[9];  mm3(w, s2, ws2);  mm3(s2, w, s2w);
#pragma unroll
  for (int i = 0; i < 9; ++i) Q[i] += g[4] * (ws2[i] - s2w[i]);

  float w2s[9], sw2[9];  mm3(w2, s, w2s);  mm3(s, w2, sw2);
#pragma unroll
  for (int i = 0; i < 9; ++i) Q[i] += g[5] * (w2s[i] + sw2[i]);
  diag += g[5] * (2.f / 3.f) * tr3(sw2);

  float wsw2[9], w2sw[9]; mm3(ws, w2, wsw2); mm3(w2, sw, w2sw);
#pragma unroll
  for (int i = 0; i < 9; ++i) Q[i] += g[6] * (wsw2[i] - w2sw[i]);

  float sws2[9]; mm3(sw, s2, sws2);           // T8 = sw@s2 - s2@w
#pragma unroll
  for (int i = 0; i < 9; ++i) Q[i] += g[7] * (sws2[i] - s2w[i]);

  float w2s2[9], s2w2[9]; mm3(w2, s2, w2s2); mm3(s2, w2, s2w2);
#pragma unroll
  for (int i = 0; i < 9; ++i) Q[i] += g[8] * (w2s2[i] + s2w2[i]);
  diag += g[8] * (2.f / 3.f) * tr3(s2w2);

  float t10a[9], t10b[9]; mm3(ws2, w2, t10a); mm3(w2, s2w, t10b);
#pragma unroll
  for (int i = 0; i < 9; ++i) Q[i] += g[9] * (t10a[i] - t10b[i]);

  Q[0] -= diag; Q[4] -= diag; Q[8] -= diag;
  const float trq = tr3(Q) * (1.f / 3.f);
  Q[0] -= trq; Q[4] -= trq; Q[8] -= trq;

  float ss = 1e-8f;
#pragma unroll
  for (int i = 0; i < 9; ++i) ss += Q[i] * Q[i];
  const float inv = 1.f / sqrtf(ss);

  if (item < nItems) {
#pragma unroll
    for (int i = 0; i < 9; ++i) out[item * 9 + i] = Q[i] * inv;
  }
}

extern "C" void kernel_launch(void* const* d_in, const int* in_sizes, int n_in,
                              void* d_out, int out_size, void* d_ws, size_t ws_size,
                              hipStream_t stream) {
  (void)n_in; (void)out_size;
  const float* S  = (const float*)d_in[0];
  const float* Wm = (const float*)d_in[1];
  const float* W0 = (const float*)d_in[2];  const float* b0 = (const float*)d_in[3];
  const float* W1 = (const float*)d_in[4];  const float* b1 = (const float*)d_in[5];
  const float* W2 = (const float*)d_in[6];  const float* b2 = (const float*)d_in[7];
  const float* W3 = (const float*)d_in[8];  const float* b3 = (const float*)d_in[9];
  const float* W4 = (const float*)d_in[10]; const float* b4 = (const float*)d_in[11];
  const float* W5 = (const float*)d_in[12]; const float* b5 = (const float*)d_in[13];

  const int nItems = in_sizes[0] / 9;          // B = 1048576
  const int blocks = (nItems + 255) / 256;
  const bool useWs = ws_size >= (size_t)W_BYTES;  // constant across calls -> deterministic

  if (useWs) {
    _Float16* wsW = (_Float16*)d_ws;
    prep_weights_kernel<<<(W_HALFS + 255) / 256, 256, 0, stream>>>(
        W0, b0, W1, b1, W2, b2, W3, b3, W4, b4, W5, b5, wsW);
    (void)hipFuncSetAttribute((const void*)tbnn_q_kernel<true>,
                              hipFuncAttributeMaxDynamicSharedMemorySize, SMEM_BYTES);
    tbnn_q_kernel<true><<<blocks, 256, SMEM_BYTES, stream>>>(
        S, Wm, W0, b0, W1, b1, W2, b2, W3, b3, W4, b4, W5, b5,
        wsW, (float*)d_out, nItems);
  } else {
    (void)hipFuncSetAttribute((const void*)tbnn_q_kernel<false>,
                              hipFuncAttributeMaxDynamicSharedMemorySize, SMEM_BYTES);
    tbnn_q_kernel<false><<<blocks, 256, SMEM_BYTES, stream>>>(
        S, Wm, W0, b0, W1, b1, W2, b2, W3, b3, W4, b4, W5, b5,
        (const _Float16*)nullptr, (float*)d_out, nItems);
  }
}
